// SWAVerifyChunk1_78159814852658
// MI455X (gfx1250) — compile-verified, matrix operand order checked by reading, not run on hardware
//
#include <hip/hip_runtime.h>
#include <hip/hip_bf16.h>
#include <math.h>

#define SEQ 128
#define HID 2048
#define NH 8
#define NKV 2
#define HD 256
#define FF 4096
#define PLD 256
#define NLT 35
#define NLAYERS 8
#define WIN 512
#define FULL_LEN 2048
#define PAST 1024

typedef __attribute__((ext_vector_type(16))) __bf16 v16bf;
typedef __attribute__((ext_vector_type(8)))  float  v8f;

__device__ __forceinline__ __bf16 f2bf(float x) { return (__bf16)x; }

// ---------------------------------------------------------------------------
// Batched GEMM: C = A(MxK) @ B(KxN), fp32 in/out, bf16 WMMA core (f32 acc).
// Each wave owns a 16-column strip and computes MT m-tiles (16 rows each),
// so the B (weight) fragment is loaded ONCE per k-step and reused by MT
// WMMAs -> HBM weight traffic multiplier is (M/16)/MT instead of M/16.
// BT=true: B stored [N,K] row-major (A @ B^T), contiguous per-lane K loads.
// Fragment layouts per CDNA5 ISA 7.12.2 (16-bit A 16x32 / B 32x16):
//   lanes 0-15  hold K = {k0..k0+7, k0+16..k0+23}
//   lanes 16-31 hold K = {k0+8..k0+15, k0+24..k0+31}
// C/D: vgpr r, lanes 0-15 -> M=r, lanes 16-31 -> M=r+8.
// ---------------------------------------------------------------------------
template<bool BT, int MT>
__global__ __launch_bounds__(128)
void gemm_bf16_wmma(const float* __restrict__ A, const float* __restrict__ B,
                    float* __restrict__ C, int M, int N, int K,
                    int lda, int ldb, int ldc,
                    long sA, long sB, long sC, int zshiftB)
{
  const int wave = threadIdx.x >> 5;
  const int lane = threadIdx.x & 31;
  const int sub  = lane & 15;
  const int hsel = lane >> 4;
  const int n0 = blockIdx.x * 64 + wave * 16;
  const int m0 = blockIdx.y * (16 * MT);
  if (n0 >= N || m0 >= M) return;           // wave-uniform; EXEC stays all-ones
  const int z = blockIdx.z;
  const float* Ab = A + (long)z * sA + (size_t)(m0 + sub) * lda;
  const float* Bb = B + (long)(z >> zshiftB) * sB;
  float*       Cb = C + (long)z * sC;
  const float* brow = BT ? (Bb + (size_t)(n0 + sub) * ldb) : Bb;

  v8f acc[MT];
#pragma unroll
  for (int t = 0; t < MT; ++t)
    acc[t] = (v8f){0.f, 0.f, 0.f, 0.f, 0.f, 0.f, 0.f, 0.f};

  for (int k0 = 0; k0 < K; k0 += 32) {
    // ---- B fragment: loaded once, reused by MT WMMAs ----
    v16bf bfm;
    if (BT) {
      __builtin_prefetch(brow + k0 + 128, 0, 1);     // global_prefetch_b8
#pragma unroll
      for (int g = 0; g < 2; ++g) {
        const float4* pb = reinterpret_cast<const float4*>(brow + k0 + g * 16 + hsel * 8);
        float4 b0 = pb[0], b1 = pb[1];
        bfm[g*8+0] = f2bf(b0.x); bfm[g*8+1] = f2bf(b0.y);
        bfm[g*8+2] = f2bf(b0.z); bfm[g*8+3] = f2bf(b0.w);
        bfm[g*8+4] = f2bf(b1.x); bfm[g*8+5] = f2bf(b1.y);
        bfm[g*8+6] = f2bf(b1.z); bfm[g*8+7] = f2bf(b1.w);
      }
    } else {
      __builtin_prefetch(Bb + (size_t)(k0 + 32) * ldb + n0 + sub, 0, 1);
#pragma unroll
      for (int g = 0; g < 2; ++g) {
        const int kk = k0 + g * 16 + hsel * 8;
#pragma unroll
        for (int t = 0; t < 8; ++t)
          bfm[g*8+t] = f2bf(Bb[(size_t)(kk + t) * ldb + n0 + sub]);
      }
    }
    // ---- MT A fragments + WMMAs ----
#pragma unroll
    for (int t = 0; t < MT; ++t) {
      const float* ar = Ab + (size_t)(t * 16) * lda;
      v16bf af;
#pragma unroll
      for (int g = 0; g < 2; ++g) {
        const float4* pa = reinterpret_cast<const float4*>(ar + k0 + g * 16 + hsel * 8);
        float4 a0 = pa[0], a1 = pa[1];
        af[g*8+0] = f2bf(a0.x); af[g*8+1] = f2bf(a0.y);
        af[g*8+2] = f2bf(a0.z); af[g*8+3] = f2bf(a0.w);
        af[g*8+4] = f2bf(a1.x); af[g*8+5] = f2bf(a1.y);
        af[g*8+6] = f2bf(a1.z); af[g*8+7] = f2bf(a1.w);
      }
      acc[t] = __builtin_amdgcn_wmma_f32_16x16x32_bf16(false, af, false, bfm,
                                                       (short)0, acc[t], false, false);
    }
  }
#pragma unroll
  for (int t = 0; t < MT; ++t)
#pragma unroll
    for (int r = 0; r < 8; ++r)
      Cb[(size_t)(m0 + t * 16 + hsel * 8 + r) * ldc + (n0 + sub)] = acc[t][r];
}

// ---------------------------------------------------------------------------
// RMS norm (wave32 reduction). y[row] = (x[row]*prescale * rsqrt(mean+eps))
//   * (w?)  (+ res[row]?)  * (*scale?)
// ---------------------------------------------------------------------------
__global__ __launch_bounds__(256)
void rms_kernel(const float* __restrict__ x, const float* __restrict__ w,
                const float* __restrict__ res, const float* __restrict__ scale,
                float* __restrict__ y, int dim, float prescale)
{
  const int row = blockIdx.x;
  const float* xr = x + (size_t)row * dim;
  float ss = 0.f;
  for (int c = threadIdx.x; c < dim; c += blockDim.x) {
    float v = xr[c] * prescale; ss += v * v;
  }
#pragma unroll
  for (int off = 16; off; off >>= 1) ss += __shfl_xor(ss, off, 32);
  __shared__ float sred[8];
  const int wid = threadIdx.x >> 5, lane = threadIdx.x & 31;
  if (lane == 0) sred[wid] = ss;
  __syncthreads();
  if (threadIdx.x == 0) {
    float t = 0.f;
    for (int i = 0; i < 8; ++i) t += sred[i];
    sred[0] = t;
  }
  __syncthreads();
  const float inv = rsqrtf(sred[0] / (float)dim + 1e-6f);
  const float sc  = scale ? *scale : 1.f;
  for (int c = threadIdx.x; c < dim; c += blockDim.x) {
    float v = xr[c] * prescale * inv;
    if (w)   v *= w[c];
    if (res) v += res[(size_t)row * dim + c];
    y[(size_t)row * dim + c] = v * sc;
  }
}

// ---------------------------------------------------------------------------
// Row-wise softmax with additive mask (scores layout [NH, SEQ, L]).
// ---------------------------------------------------------------------------
__global__ __launch_bounds__(256)
void softmax_mask_kernel(float* __restrict__ sc, const float* __restrict__ mask, int L)
{
  const int s = blockIdx.x, h = blockIdx.y;
  float* row = sc + ((size_t)h * SEQ + s) * L;
  const float* mr = mask + (size_t)s * L;
  __shared__ float sred[8];
  const int wid = threadIdx.x >> 5, lane = threadIdx.x & 31;

  float mx = -3.0e38f;
  for (int c = threadIdx.x; c < L; c += blockDim.x) mx = fmaxf(mx, row[c] + mr[c]);
#pragma unroll
  for (int off = 16; off; off >>= 1) mx = fmaxf(mx, __shfl_xor(mx, off, 32));
  if (lane == 0) sred[wid] = mx;
  __syncthreads();
  if (threadIdx.x == 0) {
    float t = sred[0];
    for (int i = 1; i < 8; ++i) t = fmaxf(t, sred[i]);
    sred[0] = t;
  }
  __syncthreads();
  mx = sred[0];
  __syncthreads();

  float sum = 0.f;
  for (int c = threadIdx.x; c < L; c += blockDim.x) {
    float e = __expf(row[c] + mr[c] - mx);
    row[c] = e; sum += e;
  }
#pragma unroll
  for (int off = 16; off; off >>= 1) sum += __shfl_xor(sum, off, 32);
  if (lane == 0) sred[wid] = sum;
  __syncthreads();
  if (threadIdx.x == 0) {
    float t = 0.f;
    for (int i = 0; i < 8; ++i) t += sred[i];
    sred[0] = t;
  }
  __syncthreads();
  const float inv = 1.f / sred[0];
  for (int c = threadIdx.x; c < L; c += blockDim.x) row[c] *= inv;
}

// ---------------------------------------------------------------------------
// RoPE in-place on [rows, HD] where rows = S*nheads; row = s*nheads + h.
// ---------------------------------------------------------------------------
__global__ __launch_bounds__(128)
void rope_kernel(float* __restrict__ x, const float* __restrict__ cs,
                 const float* __restrict__ sn, int nheads)
{
  const int row = blockIdx.x;
  const int s = row / nheads;
  float* xr = x + (size_t)row * HD;
  const int d = threadIdx.x;              // 0..127
  const float c1 = cs[s * HD + d],       s1 = sn[s * HD + d];
  const float c2 = cs[s * HD + d + 128], s2 = sn[s * HD + d + 128];
  const float x1 = xr[d], x2 = xr[d + 128];
  xr[d]       = x1 * c1 - x2 * s1;
  xr[d + 128] = x2 * c2 + x1 * s2;
}

// g[row,c] = gelu_tanh(g[row,c]) * u[row,c]  (strided u for the PLE slice)
__global__ __launch_bounds__(256)
void gelu_mul_kernel(float* __restrict__ g, int ldg,
                     const float* __restrict__ u, int ldu, int cols)
{
  const int row = blockIdx.y;
  const int c = blockIdx.x * blockDim.x + threadIdx.x;
  if (c < cols) {
    float xv = g[(size_t)row * ldg + c];
    float t = 0.7978845608028654f * (xv + 0.044715f * xv * xv * xv);
    float ge = 0.5f * xv * (1.f + tanhf(t));
    g[(size_t)row * ldg + c] = ge * u[(size_t)row * ldu + c];
  }
}

__global__ __launch_bounds__(256)
void ple_combine_kernel(const float* __restrict__ proj, const float* __restrict__ raw,
                        float* __restrict__ out, int n)
{
  const int i = blockIdx.x * blockDim.x + threadIdx.x;
  if (i < n) out[i] = (proj[i] + raw[i]) * 0.7071067811865476f;
}

// KV cache splice: dst [NKV, L, HD].
__global__ __launch_bounds__(256)
void cache_build_kernel(const float* __restrict__ oldK, const float* __restrict__ newk,
                        float* __restrict__ dst, int L, int Lold, int lo, int osh, int total)
{
  const int idx = blockIdx.x * blockDim.x + threadIdx.x;
  if (idx >= total) return;
  const int d  = idx % HD;
  const int l  = (idx / HD) % L;
  const int kv = idx / (HD * L);
  float v;
  if (l >= lo && l < lo + SEQ)
    v = newk[((size_t)(l - lo) * NKV + kv) * HD + d];
  else
    v = oldK[((size_t)kv * Lold + (l + osh)) * HD + d];
  dst[idx] = v;
}

// [S, NKV, HD] -> [NKV, S, HD]
__global__ __launch_bounds__(256)
void kv_out_kernel(const float* __restrict__ src, float* __restrict__ dst, int total)
{
  const int idx = blockIdx.x * blockDim.x + threadIdx.x;
  if (idx >= total) return;
  const int d  = idx % HD;
  const int s  = (idx / HD) % SEQ;
  const int kv = idx / (HD * SEQ);
  dst[idx] = src[((size_t)s * NKV + kv) * HD + d];
}

// ---------------------------------------------------------------------------
static inline void launch_gemm(hipStream_t st, const float* A, const float* B, float* C,
                               int M, int N, int K, int lda, int ldb, int ldc,
                               int batch, long sA, long sB, long sC, int zshiftB, bool bt)
{
  // MT m-tiles per wave: big-N GEMMs maximize weight reuse (MT=8),
  // small-N GEMMs keep launch parallelism (MT=2).
  const int mt = (N >= 4096) ? 8 : (N >= 1024 ? 4 : 2);
  dim3 g((unsigned)(N / 64), (unsigned)(M / (16 * mt)), (unsigned)batch), b(128);
  if (bt) {
    if (mt == 4)      gemm_bf16_wmma<true, 4><<<g, b, 0, st>>>(A, B, C, M, N, K, lda, ldb, ldc, sA, sB, sC, zshiftB);
    else              gemm_bf16_wmma<true, 2><<<g, b, 0, st>>>(A, B, C, M, N, K, lda, ldb, ldc, sA, sB, sC, zshiftB);
  } else {
    if (mt == 8)      gemm_bf16_wmma<false, 8><<<g, b, 0, st>>>(A, B, C, M, N, K, lda, ldb, ldc, sA, sB, sC, zshiftB);
    else if (mt == 4) gemm_bf16_wmma<false, 4><<<g, b, 0, st>>>(A, B, C, M, N, K, lda, ldb, ldc, sA, sB, sC, zshiftB);
    else              gemm_bf16_wmma<false, 2><<<g, b, 0, st>>>(A, B, C, M, N, K, lda, ldb, ldc, sA, sB, sC, zshiftB);
  }
}

extern "C" void kernel_launch(void* const* d_in, const int* in_sizes, int n_in,
                              void* d_out, int out_size, void* d_ws, size_t ws_size,
                              hipStream_t stream)
{
  (void)in_sizes; (void)n_in; (void)out_size; (void)ws_size;
  const float* hidden = (const float*)d_in[0];
  const float* plraw  = (const float*)d_in[1];
  const float* cos_s  = (const float*)d_in[2];
  const float* sin_s  = (const float*)d_in[3];
  const float* cos_f  = (const float*)d_in[4];
  const float* sin_f  = (const float*)d_in[5];
  const float* mfull  = (const float*)d_in[6];
  const float* mslid  = (const float*)d_in[7];
  /* d_in[8] update_indicator: replaced by direct cache indexing */
  const float* Ksl    = (const float*)d_in[9];
  const float* Vsl    = (const float*)d_in[10];
  const float* Kfu    = (const float*)d_in[11];
  const float* Vfu    = (const float*)d_in[12];
  const float* Wq     = (const float*)d_in[13];
  const float* Wk     = (const float*)d_in[14];
  const float* Wv     = (const float*)d_in[15];
  const float* Wo     = (const float*)d_in[16];
  const float* w_in   = (const float*)d_in[17];
  const float* w_qn   = (const float*)d_in[18];
  const float* w_kn   = (const float*)d_in[19];
  const float* w_pa   = (const float*)d_in[20];
  const float* w_pf   = (const float*)d_in[21];
  const float* Wg     = (const float*)d_in[22];
  const float* Wu     = (const float*)d_in[23];
  const float* Wd     = (const float*)d_in[24];
  const float* w_pff  = (const float*)d_in[25];
  const float* Wpg    = (const float*)d_in[26];
  const float* Wpp    = (const float*)d_in[27];
  const float* w_pp   = (const float*)d_in[28];
  const float* lsc    = (const float*)d_in[29];
  const float* Wpm    = (const float*)d_in[30];
  const float* w_pn   = (const float*)d_in[31];

  float* out = (float*)d_out;
  float* ws  = (float*)d_ws;

  size_t o = 0;
  float* hs = ws + o; o += (size_t)SEQ * HID;
  float* hn = ws + o; o += (size_t)SEQ * HID;
  float* qb = ws + o; o += (size_t)SEQ * NH * HD;
  float* kb = ws + o; o += (size_t)SEQ * NKV * HD;
  float* vb = ws + o; o += (size_t)SEQ * NKV * HD;
  float* Kc = ws + o; o += (size_t)NKV * FULL_LEN * HD;
  float* Vc = ws + o; o += (size_t)NKV * FULL_LEN * HD;
  float* sb = ws + o; o += (size_t)NH * SEQ * FULL_LEN;
  float* ab = ws + o; o += (size_t)SEQ * NH * HD;
  float* t0 = ws + o; o += (size_t)SEQ * HID;
  float* fg = ws + o; o += (size_t)SEQ * FF;
  float* fu = ws + o; o += (size_t)SEQ * FF;
  float* pl = ws + o; o += (size_t)SEQ * NLT * PLD;
  float* pj = ws + o; o += (size_t)SEQ * NLT * PLD;
  float* pg = ws + o; o += (size_t)SEQ * PLD;

  // ---------------- PLE precompute ----------------
  launch_gemm(stream, hidden, Wpm, pj, SEQ, NLT * PLD, HID, HID, NLT * PLD, NLT * PLD,
              1, 0, 0, 0, 0, false);
  rms_kernel<<<SEQ * NLT, 256, 0, stream>>>(pj, w_pn, nullptr, nullptr, pj, PLD,
                                            0.02209708691207961f /* HID^-0.5 */);
  {
    int n = SEQ * NLT * PLD;
    ple_combine_kernel<<<(n + 255) / 256, 256, 0, stream>>>(pj, plraw, pl, n);
  }
  hipMemcpyAsync(hs, hidden, sizeof(float) * SEQ * HID, hipMemcpyDeviceToDevice, stream);

  int si = 0;
  for (int i = 0; i < NLAYERS; ++i) {
    const bool full = (i == 5);
    const float* ct = full ? cos_f : cos_s;
    const float* st = full ? sin_f : sin_s;
    const float* Wqi = Wq  + (size_t)i * HID * NH * HD;
    const float* Wki = Wk  + (size_t)i * HID * NKV * HD;
    const float* Wvi = Wv  + (size_t)i * HID * NKV * HD;
    const float* Woi = Wo  + (size_t)i * NH * HD * HID;
    const float* Wgi = Wg  + (size_t)i * HID * FF;
    const float* Wui = Wu  + (size_t)i * HID * FF;
    const float* Wdi = Wd  + (size_t)i * FF * HID;
    const float* Wpgi = Wpg + (size_t)i * HID * PLD;
    const float* Wppi = Wpp + (size_t)i * PLD * HID;

    // input norm + QKV
    rms_kernel<<<SEQ, 256, 0, stream>>>(hs, w_in + (size_t)i * HID, nullptr, nullptr, hn, HID, 1.f);
    launch_gemm(stream, hn, Wqi, qb, SEQ, NH * HD, HID, HID, NH * HD, NH * HD, 1, 0, 0, 0, 0, false);
    launch_gemm(stream, hn, Wki, kb, SEQ, NKV * HD, HID, HID, NKV * HD, NKV * HD, 1, 0, 0, 0, 0, false);
    launch_gemm(stream, hn, Wvi, vb, SEQ, NKV * HD, HID, HID, NKV * HD, NKV * HD, 1, 0, 0, 0, 0, false);
    rms_kernel<<<SEQ * NH,  256, 0, stream>>>(qb, w_qn + (size_t)i * HD, nullptr, nullptr, qb, HD, 1.f);
    rms_kernel<<<SEQ * NKV, 256, 0, stream>>>(kb, w_kn + (size_t)i * HD, nullptr, nullptr, kb, HD, 1.f);
    rms_kernel<<<SEQ * NKV, 256, 0, stream>>>(vb, nullptr, nullptr, nullptr, vb, HD, 1.f);
    rope_kernel<<<SEQ * NH,  128, 0, stream>>>(qb, ct, st, NH);
    rope_kernel<<<SEQ * NKV, 128, 0, stream>>>(kb, ct, st, NKV);

    // emit new_ks / new_vs (transposed to [NKV, S, HD])
    {
      int tot = NKV * SEQ * HD;
      float* kd = out + (size_t)SEQ * HID + (size_t)i * tot;
      float* vd = out + (size_t)SEQ * HID + (size_t)NLAYERS * tot + (size_t)i * tot;
      kv_out_kernel<<<(tot + 255) / 256, 256, 0, stream>>>(kb, kd, tot);
      kv_out_kernel<<<(tot + 255) / 256, 256, 0, stream>>>(vb, vd, tot);
    }

    // build attention K/V (cache splice)
    const int L    = full ? FULL_LEN : WIN;
    const int Lold = full ? FULL_LEN : WIN;
    const int lo   = full ? PAST : (WIN - SEQ);
    const int osh  = full ? 0 : SEQ;
    const float* oK = full ? Kfu : (Ksl + (size_t)si * NKV * WIN * HD);
    const float* oV = full ? Vfu : (Vsl + (size_t)si * NKV * WIN * HD);
    {
      int tot = NKV * L * HD;
      cache_build_kernel<<<(tot + 255) / 256, 256, 0, stream>>>(oK, kb, Kc, L, Lold, lo, osh, tot);
      cache_build_kernel<<<(tot + 255) / 256, 256, 0, stream>>>(oV, vb, Vc, L, Lold, lo, osh, tot);
    }

    // attention: scores = q @ Kc^T (batched over NH, GQA via z>>2), softmax, @ Vc
    launch_gemm(stream, qb, Kc, sb, SEQ, L, HD, NH * HD, HD, L,
                NH, (long)HD, (long)L * HD, (long)SEQ * L, 2, true);
    softmax_mask_kernel<<<dim3(SEQ, NH), 256, 0, stream>>>(sb, full ? mfull : mslid, L);
    launch_gemm(stream, sb, Vc, ab, SEQ, HD, L, L, HD, NH * HD,
                NH, (long)SEQ * L, (long)L * HD, (long)HD, 2, false);

    // output projection + residual
    launch_gemm(stream, ab, Woi, t0, SEQ, HID, NH * HD, NH * HD, HID, HID, 1, 0, 0, 0, 0, false);
    rms_kernel<<<SEQ, 256, 0, stream>>>(t0, w_pa + (size_t)i * HID, hs, nullptr, hs, HID, 1.f);

    // FFN
    rms_kernel<<<SEQ, 256, 0, stream>>>(hs, w_pf + (size_t)i * HID, nullptr, nullptr, hn, HID, 1.f);
    launch_gemm(stream, hn, Wgi, fg, SEQ, FF, HID, HID, FF, FF, 1, 0, 0, 0, 0, false);
    launch_gemm(stream, hn, Wui, fu, SEQ, FF, HID, HID, FF, FF, 1, 0, 0, 0, 0, false);
    gelu_mul_kernel<<<dim3(FF / 256, SEQ), 256, 0, stream>>>(fg, FF, fu, FF, FF);
    launch_gemm(stream, fg, Wdi, t0, SEQ, HID, FF, FF, HID, HID, 1, 0, 0, 0, 0, false);
    rms_kernel<<<SEQ, 256, 0, stream>>>(t0, w_pff + (size_t)i * HID, hs, nullptr, hs, HID, 1.f);

    // PLE
    launch_gemm(stream, hs, Wpgi, pg, SEQ, PLD, HID, HID, PLD, PLD, 1, 0, 0, 0, 0, false);
    gelu_mul_kernel<<<dim3(PLD / 256, SEQ), 256, 0, stream>>>(pg, PLD, pl + (size_t)i * PLD, NLT * PLD, PLD);
    launch_gemm(stream, pg, Wppi, t0, SEQ, HID, PLD, PLD, HID, HID, 1, 0, 0, 0, 0, false);
    rms_kernel<<<SEQ, 256, 0, stream>>>(t0, w_pp + (size_t)i * HID, hs, lsc + i, hs, HID, 1.f);

    if (!full) ++si;
  }

  hipMemcpyAsync(out, hs, sizeof(float) * SEQ * HID, hipMemcpyDeviceToDevice, stream);
}
